// KnowledgeFuseBlock_38843684225075
// MI455X (gfx1250) — compile-verified, matrix operand order checked by reading, not run on hardware
//
#include <hip/hip_runtime.h>
#include <hip/hip_bf16.h>
#include <math.h>

// ---------------- problem constants (match reference) ----------------
#define LN    4096   // N = K = M = 4096 (rows of B / knowledge / ins_space)
#define DDIM  512
#define HC    256
#define HK    256
#define NSEL  128    // n_ins
#define NBAG  128
#define BAGSTEP 32   // M / NBAG
#define FUSEW 768    // D + n_ins + n_bag

// ---------------- GEMM tiling ----------------
#define BM 128
#define BN 128
#define KT 32
#define LDK (KT + 2)   // bf16 row pad -> conflict-free fragment reads

typedef __attribute__((ext_vector_type(16))) __bf16 v16bf;
typedef __attribute__((ext_vector_type(8)))  float  v8f;

__device__ __forceinline__ unsigned short f2bf(float x) {
  unsigned u = __builtin_bit_cast(unsigned, x);
  return (unsigned short)((u + 0x7FFFu + ((u >> 16) & 1u)) >> 16); // RNE
}
__device__ __forceinline__ unsigned packbf(float x, float y) {
  return (unsigned)f2bf(x) | ((unsigned)f2bf(y) << 16);
}

// Epilogues: 0 none | 1 bias+lrelu | 2 bias+lrelu+BN(g,be) | 3 bias+tanh
//            4 +residual | 5 cdist(rownorm,colnorm)
// TA=0: A is [M][K] row-major (lda=row stride). TA=1: A is [K][M] (lda=row stride).
// TB=1: B is [N][K] (NT).                        TB=0: B is [K][N] (NN).
// Block tile 128x128, 8 waves as 4(M) x 2(N); each wave computes 32x64
// (2 A-frags x 4 B-frags -> 8 WMMAs per stage).
template<int TA, int TB>
__global__ void __launch_bounds__(256) wmma_gemm(
    const float* __restrict__ A, const float* __restrict__ Bm, float* __restrict__ C,
    int M, int N, int K, int lda, int ldb, int ldc, int epi,
    const float* __restrict__ p0, const float* __restrict__ p1,
    const float* __restrict__ p2)
{
  __shared__ unsigned short As[BM][LDK];
  __shared__ unsigned short Bs[BN][LDK];
  const int tid  = threadIdx.x;
  const int lane = tid & 31;
  const int wave = tid >> 5;       // 0..7
  const int wrow = wave >> 1;      // 0..3  (32-row strip)
  const int wcol = wave & 1;       // 0..1  (64-col strip)
  const int half = lane >> 4;
  const int l16  = lane & 15;
  const int bm = blockIdx.x * BM;
  const int bn = blockIdx.y * BN;

  v8f acc[2][4];
#pragma unroll
  for (int a = 0; a < 2; a++)
#pragma unroll
    for (int t = 0; t < 4; t++)
#pragma unroll
      for (int r = 0; r < 8; r++) acc[a][t][r] = 0.0f;

  for (int k0 = 0; k0 < K; k0 += KT) {
    // ---- stage A tile (BM x KT) as bf16 ----
    if (TA == 0) {
#pragma unroll
      for (int i = 0; i < 8; i++) {            // 128*16 pairs / 256 thr
        int e  = tid + 256 * i;
        int r  = e >> 4;
        int kp = (e & 15) << 1;
        const float* s = A + (size_t)(bm + r) * lda + (k0 + kp);
        *(unsigned*)&As[r][kp] = packbf(s[0], s[1]);
      }
    } else {                                    // transposed staging
#pragma unroll
      for (int i = 0; i < 16; i++) {           // 128*32 elems / 256 thr
        int e  = tid + 256 * i;
        int m  = e & (BM - 1);
        int kk = e >> 7;
        As[m][kk] = f2bf(A[(size_t)(k0 + kk) * lda + bm + m]);
      }
    }
    // ---- stage B tile (BN x KT) as bf16 ----
    if (TB == 1) {
#pragma unroll
      for (int i = 0; i < 8; i++) {            // 128*16 pairs / 256 thr
        int e  = tid + 256 * i;
        int r  = e >> 4;
        int kp = (e & 15) << 1;
        const float* s = Bm + (size_t)(bn + r) * ldb + (k0 + kp);
        *(unsigned*)&Bs[r][kp] = packbf(s[0], s[1]);
      }
    } else {
#pragma unroll
      for (int i = 0; i < 16; i++) {           // 128*32 elems / 256 thr
        int e  = tid + 256 * i;
        int n  = e & (BN - 1);
        int kk = e >> 7;
        Bs[n][kk] = f2bf(Bm[(size_t)(k0 + kk) * ldb + bn + n]);
      }
    }
    __syncthreads();

    // ---- fragments per ISA 16-bit 16x32 VGPR layout ----
    union { unsigned u[8]; v16bf v; } fa[2];
#pragma unroll
    for (int a = 0; a < 2; a++)
#pragma unroll
      for (int p = 0; p < 8; p++) {
        int kb = ((p >> 2) << 4) + (half << 3) + ((p & 3) << 1); // pair base
        fa[a].u[p] = *(const unsigned*)&As[(wrow << 5) + (a << 4) + l16][kb];
      }
#pragma unroll
    for (int t = 0; t < 4; t++) {
      union { unsigned u[8]; v16bf v; } fb;
#pragma unroll
      for (int p = 0; p < 8; p++) {
        int kb = ((p >> 2) << 4) + (half << 3) + ((p & 3) << 1);
        fb.u[p] = *(const unsigned*)&Bs[(wcol << 6) + (t << 4) + l16][kb];
      }
#pragma unroll
      for (int a = 0; a < 2; a++)
        acc[a][t] = __builtin_amdgcn_wmma_f32_16x16x32_bf16(
            false, fa[a].v, false, fb.v, (short)0, acc[a][t], false, false);
    }
    __syncthreads();
  }

  // ---- epilogue + store (C/D layout: row = r + 8*half, col = l16) ----
  const float bninv = 0.9999950000374997f; // 1/sqrt(1+1e-5)
#pragma unroll
  for (int a = 0; a < 2; a++) {
#pragma unroll
    for (int t = 0; t < 4; t++) {
      int col = bn + (wcol << 6) + (t << 4) + l16;
      if (col >= N) continue;
#pragma unroll
      for (int r = 0; r < 8; r++) {
        int row = bm + (wrow << 5) + (a << 4) + r + (half << 3);
        if (row >= M) continue;
        float v = acc[a][t][r];
        if (epi == 1)      { v += p0[col]; v = v > 0.f ? v : 0.01f * v; }
        else if (epi == 2) { v += p0[col]; v = v > 0.f ? v : 0.01f * v;
                             v = v * (p1[col] * bninv) + p2[col]; }
        else if (epi == 3) { v += p0[col]; v = tanhf(v); }
        else if (epi == 4) { v += p0[(size_t)row * ldc + col]; }
        else if (epi == 5) { float d = p0[row] + p1[col] - 2.0f * v;
                             v = sqrtf(fmaxf(d, 0.0f)); }
        C[(size_t)row * ldc + col] = v;
      }
    }
  }
}

// softmax over axis 1 (down the rows, per column), in place
__global__ void col_softmax(float* __restrict__ F, int rows, int cols) {
  int j = blockIdx.x * blockDim.x + threadIdx.x;
  if (j >= cols) return;
  float mx = -3.4e38f;
  for (int i = 0; i < rows; i++) mx = fmaxf(mx, F[(size_t)i * cols + j]);
  float s = 0.f;
  for (int i = 0; i < rows; i++) s += expf(F[(size_t)i * cols + j] - mx);
  float inv = 1.0f / s;
  for (int i = 0; i < rows; i++) {
    size_t o = (size_t)i * cols + j;
    F[o] = expf(F[o] - mx) * inv;
  }
}

__global__ void row_sqnorm(const float* __restrict__ X, float* __restrict__ out, int cols) {
  __shared__ float red[256];
  int i = blockIdx.x;
  float s = 0.f;
  for (int j = threadIdx.x; j < cols; j += blockDim.x) {
    float v = X[(size_t)i * cols + j]; s += v * v;
  }
  red[threadIdx.x] = s; __syncthreads();
  for (int st = 128; st > 0; st >>= 1) {
    if ((int)threadIdx.x < st) red[threadIdx.x] += red[threadIdx.x + st];
    __syncthreads();
  }
  if (threadIdx.x == 0) out[i] = red[0];
}

// logits[m] = sum_h T[h][m]*w4[h] + b4[0]   (T is [H][dm] row-major)
__global__ void mask_logits(const float* __restrict__ T, const float* __restrict__ w4,
                            const float* __restrict__ b4, float* __restrict__ out,
                            int H, int dm) {
  int m = blockIdx.x * blockDim.x + threadIdx.x;
  if (m >= dm) return;
  float s = b4[0];
  for (int h = 0; h < H; h++) s += T[(size_t)h * dm + m] * w4[h];
  out[m] = s;
}

// stable top-128 (descending value, ascending index on ties) over 4096 scores
__global__ void topk128(const float* __restrict__ score, int* __restrict__ idx) {
  __shared__ float vals[4096];
  __shared__ float bv[256];
  __shared__ int   bi[256];
  for (int i = threadIdx.x; i < 4096; i += 256) vals[i] = score[i];
  __syncthreads();
  for (int t = 0; t < NSEL; t++) {
    float best = -3.4e38f; int besti = 4096;
    for (int i = threadIdx.x; i < 4096; i += 256) {
      float v = vals[i];
      if (v > best || (v == best && i < besti)) { best = v; besti = i; }
    }
    bv[threadIdx.x] = best; bi[threadIdx.x] = besti;
    __syncthreads();
    if (threadIdx.x == 0) {
      float Bv = -3.4e38f; int Bi = 4096;
      for (int k = 0; k < 256; k++)
        if (bv[k] > Bv || (bv[k] == Bv && bi[k] < Bi)) { Bv = bv[k]; Bi = bi[k]; }
      idx[t] = Bi; vals[Bi] = -3.4e38f;
    }
    __syncthreads();
  }
}

// cols[b] = argmax(score[b*step:(b+1)*step]) + ins_idx[b]  (first max wins)
__global__ void bag_cols(const float* __restrict__ score, const int* __restrict__ ins_idx,
                         int* __restrict__ cols) {
  int b = threadIdx.x;
  if (b >= NBAG) return;
  int base = b * BAGSTEP;
  float best = score[base]; int bi = 0;
  for (int s = 1; s < BAGSTEP; s++) {
    float v = score[base + s];
    if (v > best) { best = v; bi = s; }
  }
  cols[b] = bi + ins_idx[b];
}

// row-wise softmax over 128 gathered columns, written into fuse[:, coff:coff+128]
__global__ void gather_softmax(const float* __restrict__ Aff, const int* __restrict__ sel,
                               float* __restrict__ fuse, int srcCols, int ldf, int coff) {
  __shared__ float red[128];
  int i = blockIdx.x;
  int c = threadIdx.x;                      // 128 threads
  float v = Aff[(size_t)i * srcCols + sel[c]];
  red[c] = v; __syncthreads();
  for (int st = 64; st > 0; st >>= 1) {
    if (c < st) red[c] = fmaxf(red[c], red[c + st]);
    __syncthreads();
  }
  float mx = red[0]; __syncthreads();
  float e = expf(v - mx);
  red[c] = e; __syncthreads();
  for (int st = 64; st > 0; st >>= 1) {
    if (c < st) red[c] += red[c + st];
    __syncthreads();
  }
  float s = red[0];
  fuse[(size_t)i * ldf + coff + c] = e / s;
}

__global__ void copy_into_fuse(const float* __restrict__ B, float* __restrict__ fuse,
                               int rows, int d, int ldf) {
  int e = blockIdx.x * blockDim.x + threadIdx.x;
  if (e >= rows * d) return;
  int i = e / d, j = e % d;
  fuse[(size_t)i * ldf + j] = B[(size_t)i * d + j];
}

// ---------------- host side ----------------
struct SkipP { const float *w1,*b1,*g1,*be1,*w2,*b2,*g2,*be2,*w3,*b3,*g3,*be3; };
struct MaskP { const float *W1,*b1,*W2,*b2,*W3,*b3,*w4,*b4; };

static SkipP mkSkip(void* const* d, int b) {
  SkipP s;
  s.w1=(const float*)d[b+0];  s.b1=(const float*)d[b+1];  s.g1=(const float*)d[b+2];  s.be1=(const float*)d[b+3];
  s.w2=(const float*)d[b+4];  s.b2=(const float*)d[b+5];  s.g2=(const float*)d[b+6];  s.be2=(const float*)d[b+7];
  s.w3=(const float*)d[b+8];  s.b3=(const float*)d[b+9];  s.g3=(const float*)d[b+10]; s.be3=(const float*)d[b+11];
  return s;
}
static MaskP mkMask(void* const* d, int b) {
  MaskP m;
  m.W1=(const float*)d[b+0]; m.b1=(const float*)d[b+1];
  m.W2=(const float*)d[b+2]; m.b2=(const float*)d[b+3];
  m.W3=(const float*)d[b+4]; m.b3=(const float*)d[b+5];
  m.w4=(const float*)d[b+6]; m.b4=(const float*)d[b+7];
  return m;
}

static void gemm(hipStream_t st, int TA, int TB,
                 const float* A, const float* B, float* C,
                 int M, int N, int K, int lda, int ldb, int ldc, int epi,
                 const float* p0 = nullptr, const float* p1 = nullptr,
                 const float* p2 = nullptr) {
  dim3 g((M + BM - 1) / BM, (N + BN - 1) / BN), blk(256);
  if (TA == 0 && TB == 1)
    hipLaunchKernelGGL((wmma_gemm<0,1>), g, blk, 0, st, A,B,C,M,N,K,lda,ldb,ldc,epi,p0,p1,p2);
  else if (TA == 0 && TB == 0)
    hipLaunchKernelGGL((wmma_gemm<0,0>), g, blk, 0, st, A,B,C,M,N,K,lda,ldb,ldc,epi,p0,p1,p2);
  else
    hipLaunchKernelGGL((wmma_gemm<1,0>), g, blk, 0, st, A,B,C,M,N,K,lda,ldb,ldc,epi,p0,p1,p2);
}

// SkipConnectBlock: out = colsoftmax(X1@X2^T) @ X3 + X
static void run_skip(hipStream_t st, const float* X, int L, int din, const SkipP& p,
                     float* out, float* x1, float* x2, float* x3, float* feat) {
  gemm(st, 0,1, X, p.w1, x1,  L, HC,  din, din, din, HC,  2, p.b1, p.g1, p.be1);
  gemm(st, 0,1, X, p.w2, x2,  L, HC,  din, din, din, HC,  2, p.b2, p.g2, p.be2);
  gemm(st, 0,1, X, p.w3, x3,  L, din, din, din, din, din, 2, p.b3, p.g3, p.be3);
  gemm(st, 0,1, x1, x2, feat, L, L,   HC,  HC,  HC,  L,   0);
  hipLaunchKernelGGL(col_softmax, dim3(L/256), dim3(256), 0, st, feat, L, L);
  gemm(st, 0,0, feat, x3, out, L, din, L,  L,   din, din, 4, X);
}

// MaskBlock -> attention logits (softmax skipped: selection is monotone-invariant)
static void run_mask(hipStream_t st, const float* affs, int L, int dm, const MaskP& p,
                     float* x1, float* x2, float* amat, float* maskT, float* logits) {
  gemm(st, 0,1, affs, p.W1, x1, L, HK, dm, dm, dm, HK, 1, p.b1);
  gemm(st, 0,1, affs, p.W2, x2, L, HK, dm, dm, dm, HK, 3, p.b2);
  gemm(st, 1,0, x1, x2, amat, HK, HK, L, HK, HK, HK, 0);           // X1^T @ X2
  gemm(st, 0,1, amat, p.W3, maskT, HK, dm, HK, HK, HK, dm, 1, p.b3);
  hipLaunchKernelGGL(mask_logits, dim3(dm/256), dim3(256), 0, st,
                     maskT, p.w4, p.b4, logits, HK, dm);
}

extern "C" void kernel_launch(void* const* d_in, const int* in_sizes, int n_in,
                              void* d_out, int out_size, void* d_ws, size_t ws_size,
                              hipStream_t stream) {
  (void)in_sizes; (void)n_in; (void)out_size; (void)ws_size;
  const float* Bx   = (const float*)d_in[0];   // [4096,512]
  const float* Kin  = (const float*)d_in[1];   // [4096,512]
  const float* Sp   = (const float*)d_in[2];   // [4096,512]
  const int*   insIdx = (const int*)d_in[3];   // [129]
  // d_in[4] = n_ins (== 128, hardcoded)
  // params flattened in dict-insertion order
  SkipP insSkip  = mkSkip(d_in, 5);
  SkipP insSkip2 = mkSkip(d_in, 17);
  SkipP insSkip3 = mkSkip(d_in, 29);
  MaskP insMask  = mkMask(d_in, 41);
  SkipP bagSkip  = mkSkip(d_in, 49);
  SkipP bagSkip2 = mkSkip(d_in, 61);
  SkipP bagSkip3 = mkSkip(d_in, 73);
  MaskP bagMask  = mkMask(d_in, 85);
  const float* embW = (const float*)d_in[93];  // [512,768]
  const float* embB = (const float*)d_in[94];  // [512]

  // workspace carve-out
  float* w = (float*)d_ws;
  size_t off = 0;
  auto take = [&](size_t n) { float* p = w + off; off += n; return p; };
  float* bufAff   = take((size_t)LN * LN);
  float* bufAffs  = take((size_t)LN * LN);
  float* bufFeat  = take((size_t)LN * LN);
  float* bufX3    = take((size_t)LN * LN);
  float* bufX1    = take((size_t)LN * HC);
  float* bufX2    = take((size_t)LN * HC);
  float* bufXs    = take((size_t)LN * DDIM);
  float* bufKs    = take((size_t)LN * DDIM);
  float* bufFuse  = take((size_t)LN * FUSEW);
  float* bufMaskT = take((size_t)HK * LN);
  float* bufAmat  = take((size_t)HK * HK);
  float* rn       = take(LN);
  float* cn       = take(LN);
  float* logits   = take(LN);
  int*   idxb     = (int*)take(128);
  int*   colsb    = (int*)take(128);

  // ===== KnowledgeInsFuseBlock =====
  run_skip(stream, Bx,  LN, DDIM, insSkip,  bufXs, bufX1, bufX2, bufX3, bufFeat);
  run_skip(stream, Kin, LN, DDIM, insSkip2, bufKs, bufX1, bufX2, bufX3, bufFeat);
  hipLaunchKernelGGL(row_sqnorm, dim3(LN), dim3(256), 0, stream, bufXs, rn, DDIM);
  hipLaunchKernelGGL(row_sqnorm, dim3(LN), dim3(256), 0, stream, bufKs, cn, DDIM);
  gemm(stream, 0,1, bufXs, bufKs, bufAff, LN, LN, DDIM, DDIM, DDIM, LN, 5, rn, cn);
  run_skip(stream, bufAff, LN, LN, insSkip3, bufAffs, bufX1, bufX2, bufX3, bufFeat);
  run_mask(stream, bufAffs, LN, LN, insMask, bufX1, bufX2, bufAmat, bufMaskT, logits);
  hipLaunchKernelGGL(topk128, dim3(1), dim3(256), 0, stream, logits, idxb);
  hipLaunchKernelGGL(gather_softmax, dim3(LN), dim3(128), 0, stream,
                     bufAffs, idxb, bufFuse, LN, FUSEW, DDIM);

  // ===== KnowledgeBagFuseBlock =====
  run_skip(stream, Bx, LN, DDIM, bagSkip,  bufXs, bufX1, bufX2, bufX3, bufFeat);
  run_skip(stream, Sp, LN, DDIM, bagSkip2, bufKs, bufX1, bufX2, bufX3, bufFeat);
  hipLaunchKernelGGL(row_sqnorm, dim3(LN), dim3(256), 0, stream, bufXs, rn, DDIM);
  hipLaunchKernelGGL(row_sqnorm, dim3(LN), dim3(256), 0, stream, bufKs, cn, DDIM);
  gemm(stream, 0,1, bufXs, bufKs, bufAff, LN, LN, DDIM, DDIM, DDIM, LN, 5, rn, cn);
  run_skip(stream, bufAff, LN, LN, bagSkip3, bufAffs, bufX1, bufX2, bufX3, bufFeat);
  run_mask(stream, bufAffs, LN, LN, bagMask, bufX1, bufX2, bufAmat, bufMaskT, logits);
  hipLaunchKernelGGL(bag_cols, dim3(1), dim3(128), 0, stream, logits, insIdx, colsb);
  hipLaunchKernelGGL(gather_softmax, dim3(LN), dim3(128), 0, stream,
                     bufAffs, colsb, bufFuse, LN, FUSEW, DDIM + NSEL);

  // ===== fuse + embedding =====
  hipLaunchKernelGGL(copy_into_fuse, dim3((LN * DDIM + 255) / 256), dim3(256), 0, stream,
                     Bx, bufFuse, LN, DDIM, FUSEW);
  gemm(stream, 0,1, bufFuse, embW, (float*)d_out, LN, DDIM, FUSEW, FUSEW, FUSEW, DDIM,
       1, embB);
}
